// DeepConsistencyLoss_32220844655142
// MI455X (gfx1250) — matrix operand order repaired and checked by reading, not compiled, not run on hardware
//
#include <hip/hip_runtime.h>
#include <hip/hip_bf16.h>
#include <math.h>

typedef __attribute__((ext_vector_type(16))) _Float16 v16h;
typedef __attribute__((ext_vector_type(8)))  _Float16 v8h;
typedef __attribute__((ext_vector_type(8)))  float    v8f;

#define IMG 512
#define NIMG 32
#define NTILES (NIMG * 32 * 32)   // 32768 16x16 tiles
#define WPB 8                     // waves per block
#define NPIX 8388608.0f           // 32*512*512

// Per-wave LDS slice (f16 throughout):
//  A,B : centered sigmoid tiles over the 22-row halo; cols 22..31 hold harmless
//        overhang (Wh rows >21 are zero, so they never contribute)
//  Ht  : horizontally-blurred tile, transposed [col][row]; ping-pong x2 so
//        consecutive channels don't serialize on the same buffer
//  WT  : 64-entry f16 tap table, WT[16+d] = w[d] for d in [0,6], else 0
struct WaveLds {
    _Float16 A[22][32];
    _Float16 B[22][32];
    _Float16 Ht[2][16][32];
    _Float16 WT[64];
};

__device__ __forceinline__ v8f wmma_f16(v16h a, v16h b, v8f c) {
    return __builtin_amdgcn_wmma_f32_16x16x32_f16(false, a, false, b, (short)0, c, false, false);
}

// sigmoid(x) - 0.5 == 0.5*tanh(x/2); gfx1250 has native v_tanh_f32
__device__ __forceinline__ float sigc(float x) {
#if __has_builtin(__builtin_amdgcn_tanhf)
    return 0.5f * __builtin_amdgcn_tanhf(0.5f * x);
#elif __has_builtin(__builtin_amdgcn_tanh_f32)
    return 0.5f * __builtin_amdgcn_tanh_f32(0.5f * x);
#else
    return 1.f / (1.f + expf(-x)) - 0.5f;
#endif
}

__device__ __forceinline__ float ssim_at(float Ma, float Mb, float Paa, float Pbb,
                                         float Pab, float K) {
    const float C1 = 0.0001f, C2 = 0.0009f;
    const float mu1 = Ma + 0.5f * K, mu2 = Mb + 0.5f * K;
    const float mu1sq = mu1 * mu1, mu2sq = mu2 * mu2, mu12 = mu1 * mu2;
    // blur(s^2) = Paa + Ma + 0.25K   (s = a + 0.5 inside the valid mask)
    const float s1  = Paa + Ma + 0.25f * K - mu1sq;
    const float s2  = Pbb + Mb + 0.25f * K - mu2sq;
    const float s12 = Pab + 0.5f * (Ma + Mb) + 0.25f * K - mu12;
    return ((2.f * mu12 + C1) * (2.f * s12 + C2)) /
           ((mu1sq + mu2sq + C1) * (s1 + s2 + C2));
}

__global__ void __launch_bounds__(WPB * 32)
ssim_tiles_wmma(const float* __restrict__ seg, const float* __restrict__ edge,
                float* __restrict__ tile_sums)
{
    __shared__ WaveLds lds[WPB];

    const int lane = threadIdx.x & 31;
    const int wave = threadIdx.x >> 5;
    const int tile = blockIdx.x * WPB + wave;

    const int n   = tile >> 10;
    const int rem = tile & 1023;
    const int y0  = (rem >> 5) << 4;
    const int x0  = (rem & 31) << 4;
    const int base = n * (IMG * IMG);

    WaveLds& L = lds[wave];

    // Gaussian taps (sigma=1.5, ws=7): compiler constant-folds these to literals
    float w[7];
    {
        float s = 0.f;
        #pragma unroll
        for (int j = 0; j < 7; ++j) { float d = (float)(j - 3); w[j] = expf(-(d*d) / 4.5f); s += w[j]; }
        #pragma unroll
        for (int j = 0; j < 7; ++j) w[j] /= s;
    }

    // ---- tap lookup table (once, 2 selects per lane)
    {
        float r0 = 0.f, r1 = 0.f;
        const int d0 = lane - 16, d1 = lane + 16;
        #pragma unroll
        for (int j = 0; j < 7; ++j) {
            r0 = (d0 == j) ? w[j] : r0;
            r1 = (d1 == j) ? w[j] : r1;
        }
        L.WT[lane]      = (_Float16)r0;
        L.WT[lane + 32] = (_Float16)r1;
    }

    // ---- Phase A: stage centered sigmoid halo tiles as f16.
    // Row-per-iteration: gy is wave-uniform (scalar branch / scalar addressing),
    // gx and its clamp/mask computed once.
    {
        const int  gx  = x0 - 3 + lane;
        const bool xok = (unsigned)gx < IMG;
        const int  gxc = min(max(gx, 0), IMG - 1);
        const float* segp = seg  + base + gxc;
        const float* edgp = edge + base + gxc;
        const _Float16 h0 = (_Float16)0.f;
        for (int r = 0; r < 22; ++r) {
            const int gy = y0 - 3 + r;
            _Float16 sv = h0, ev = h0;
            if ((unsigned)gy < IMG) {                 // uniform branch
                const int off = gy * IMG;
                const _Float16 s16 = (_Float16)sigc(segp[off]);
                const _Float16 e16 = (_Float16)sigc(edgp[off]);
                sv = xok ? s16 : h0;
                ev = xok ? e16 : h0;
            }
            L.A[r][lane] = sv;
            L.B[r][lane] = ev;
        }
    }
    // single wave per tile -> same-wave DS ordering, no barriers anywhere

    // ---- Weight fragments from the LDS tap table
    const int col = lane & 15;   // N index / A-matrix M index
    const int hh  = lane >> 4;   // lane half
    const int kb  = hh << 4;     // unused; B operands also take their k base from k8
    const int k8  = hh << 3;     // A-operand k base within 16 (0/8)

    v16h whB;                    // whB[t] = Wh[k8+t][col] = w[k8+t-col] (and +16 half)
    {
        const int bB = 16 + k8 - col;
        #pragma unroll
        for (int t = 0; t < 8; ++t) {
            whB[t]     = L.WT[bB + t];
            whB[8 + t] = L.WT[bB + 16 + t];
        }
    }
    v16h avA;                    // avA[t] = Av[col][k8+t] = w[k8+t-col] (and +16 half)
    {
        const int bA = 16 + k8 - col;
        #pragma unroll
        for (int t = 0; t < 8; ++t) {
            avA[t]     = L.WT[bA + t];
            avA[8 + t] = L.WT[bA + 16 + t];
        }
    }

    // ---- Phase B: 4 shared data fragments; product channels via v_pk_mul_f16
    v16h fa0, fa6, fb0, fb6;
    {
        v8h lo, hi;
        lo = *(const v8h*)&L.A[col][k8];     hi = *(const v8h*)&L.A[col][16 + k8];
        fa0 = __builtin_shufflevector(lo, hi, 0,1,2,3,4,5,6,7,8,9,10,11,12,13,14,15);
        lo = *(const v8h*)&L.A[col + 6][k8]; hi = *(const v8h*)&L.A[col + 6][16 + k8];
        fa6 = __builtin_shufflevector(lo, hi, 0,1,2,3,4,5,6,7,8,9,10,11,12,13,14,15);
        lo = *(const v8h*)&L.B[col][k8];     hi = *(const v8h*)&L.B[col][16 + k8];
        fb0 = __builtin_shufflevector(lo, hi, 0,1,2,3,4,5,6,7,8,9,10,11,12,13,14,15);
        lo = *(const v8h*)&L.B[col + 6][k8]; hi = *(const v8h*)&L.B[col + 6][16 + k8];
        fb6 = __builtin_shufflevector(lo, hi, 0,1,2,3,4,5,6,7,8,9,10,11,12,13,14,15);
    }
    const v16h f0[5] = { fa0, fb0, fa0 * fa0, fb0 * fb0, fa0 * fb0 };
    const v16h f6[5] = { fa6, fb6, fa6 * fa6, fb6 * fb6, fa6 * fb6 };

    v8f acc[5];
    const v8f z = {0.f, 0.f, 0.f, 0.f, 0.f, 0.f, 0.f, 0.f};
    #pragma unroll
    for (int ch = 0; ch < 5; ++ch) {
        _Float16 (*Ht)[32] = L.Ht[ch & 1];          // ping-pong

        // horizontal blur: rows 0..15 and 6..21
        v8f h0 = wmma_f16(f0[ch], whB, z);
        v8f h1 = wmma_f16(f6[ch], whB, z);

        // scatter H transposed into Ht[col][row]
        v8h p;
        #pragma unroll
        for (int v = 0; v < 8; ++v) p[v] = (_Float16)h0[v];
        *(v8h*)&Ht[col][8 * hh] = p;                // rows 0..15, one b128 store
        if (hh) {
            #pragma unroll
            for (int v = 2; v < 8; ++v)
                Ht[col][14 + v] = (_Float16)h1[v];  // rows 16..21
        }

        // vertical blur: B-fragment rows k8..k8+7 and 16+k8..16+k8+7 per lane
        v8h lo = *(const v8h*)&Ht[col][k8];
        v8h hi = *(const v8h*)&Ht[col][16 + k8];
        v16h bv = __builtin_shufflevector(lo, hi, 0,1,2,3,4,5,6,7,8,9,10,11,12,13,14,15);
        acc[ch] = wmma_f16(avA, bv, z);
    }

    // ---- Phase C: SSIM map. Interior tiles (88%) take the K==1 fast path.
    float lsum = 0.f;
    const bool border = (x0 == 0) || (y0 == 0) || (x0 == IMG - 16) || (y0 == IMG - 16);
    if (!border) {
        #pragma unroll
        for (int v = 0; v < 8; ++v)
            lsum += ssim_at(acc[0][v], acc[1][v], acc[2][v], acc[3][v], acc[4][v], 1.0f);
    } else {
        const int x = x0 + col;
        float kx = 0.f;
        #pragma unroll
        for (int j = 0; j < 7; ++j) {
            const int gx = x - 3 + j;
            if ((unsigned)gx < IMG) kx += w[j];
        }
        #pragma unroll
        for (int v = 0; v < 8; ++v) {
            const int y = y0 + v + 8 * hh;
            float ky = 0.f;
            #pragma unroll
            for (int j = 0; j < 7; ++j) {
                const int gy = y - 3 + j;
                if ((unsigned)gy < IMG) ky += w[j];
            }
            lsum += ssim_at(acc[0][v], acc[1][v], acc[2][v], acc[3][v], acc[4][v], kx * ky);
        }
    }

    // ---- wave reduction, one partial per tile (deterministic, atomic-free)
    #pragma unroll
    for (int off = 16; off > 0; off >>= 1)
        lsum += __shfl_down(lsum, off, 32);
    if (lane == 0) tile_sums[tile] = lsum;
}

__global__ void __launch_bounds__(256)
ssim_finalize(const float* __restrict__ tile_sums, float* __restrict__ out)
{
    __shared__ float red[256];
    float s = 0.f;
    for (int i = threadIdx.x; i < NTILES; i += 256) s += tile_sums[i];
    red[threadIdx.x] = s;
    __syncthreads();
    #pragma unroll
    for (int st = 128; st > 0; st >>= 1) {
        if (threadIdx.x < st) red[threadIdx.x] += red[threadIdx.x + st];
        __syncthreads();
    }
    if (threadIdx.x == 0)
        out[0] = 0.5f - 0.5f * (red[0] / NPIX);   // 1 - (1+mean)/2
}

extern "C" void kernel_launch(void* const* d_in, const int* in_sizes, int n_in,
                              void* d_out, int out_size, void* d_ws, size_t ws_size,
                              hipStream_t stream) {
    const float* seg  = (const float*)d_in[0];
    const float* edge = (const float*)d_in[1];
    float* sums = (float*)d_ws;                 // NTILES floats = 128 KB scratch
    float* out  = (float*)d_out;

    ssim_tiles_wmma<<<NTILES / WPB, WPB * 32, 0, stream>>>(seg, edge, sums);
    ssim_finalize<<<1, 256, 0, stream>>>(sums, out);
}
